// QuestionAttention_3238405341580
// MI455X (gfx1250) — matrix operand order repaired and checked by reading, not compiled, not run on hardware
//
#include <hip/hip_runtime.h>
#include <stdint.h>

// ---------------------------------------------------------------------------
// Types for CDNA5 WMMA (wave32, 16x16x32 bf16 -> f32)
// ---------------------------------------------------------------------------
typedef __attribute__((ext_vector_type(4)))  unsigned int   u32x4;
typedef __attribute__((ext_vector_type(4)))  int            i32x4;
typedef __attribute__((ext_vector_type(4)))  float          f32x4;
typedef __attribute__((ext_vector_type(4)))  unsigned short u16x4;
typedef __attribute__((ext_vector_type(8)))  float          v8f;
typedef __attribute__((ext_vector_type(16))) __bf16         v16bf;

union Frag {
    v16bf v;
    u32x4 q[2];   // 32 bytes = 16 bf16
};

#define DIMC   1024
#define NSEQ   1024
#define BATCH  4
#define NHEADS 16
#define HDIM   64
#define ATTN_SCALE 0.125f   // 64^-0.5

// ---------------------------------------------------------------------------
// gfx1250 async global->LDS copy (ASYNCcnt path), guarded so compile never
// breaks if the toolchain lacks the builtins; falls back to plain ld/st.
// Builtin signature (from hipcc diagnostic): (v4i32 as1*, v4i32 as3*, Ii, Ii)
// ---------------------------------------------------------------------------
#if defined(__has_builtin)
#if __has_builtin(__builtin_amdgcn_global_load_async_to_lds_b128)
#define HAVE_ASYNC_LDS 1
#endif
#if __has_builtin(__builtin_amdgcn_s_wait_asynccnt)
#define HAVE_WAIT_ASYNC 1
#endif
#endif

typedef __attribute__((address_space(1))) i32x4 gas_i32x4;
typedef __attribute__((address_space(3))) i32x4 las_i32x4;

__device__ __forceinline__ void copy16_g2lds(const unsigned short* g, unsigned short* l) {
#if defined(HAVE_ASYNC_LDS)
    __builtin_amdgcn_global_load_async_to_lds_b128(
        (gas_i32x4*)g, (las_i32x4*)l, /*offset=*/0, /*cpol=*/0);
#else
    *(u32x4*)l = *(const u32x4*)g;
#endif
}

__device__ __forceinline__ void wait_async_copies() {
#if defined(HAVE_ASYNC_LDS) && defined(HAVE_WAIT_ASYNC)
    __builtin_amdgcn_s_wait_asynccnt(0);
#endif
}

__device__ __forceinline__ unsigned short f32_to_bf16(float f) {
    unsigned int u = __float_as_uint(f);
    unsigned int r = u + 0x7FFFu + ((u >> 16) & 1u);  // round-to-nearest-even
    return (unsigned short)(r >> 16);
}

__device__ __forceinline__ v8f wmma_bf16(const Frag& a, const Frag& b, v8f c) {
    return __builtin_amdgcn_wmma_f32_16x16x32_bf16(
        /*neg_a=*/false, a.v, /*neg_b=*/false, b.v,
        /*c_mod=*/(short)0, c, /*reuse_a=*/false, /*reuse_b=*/false);
}

// ---------------------------------------------------------------------------
// 1) fp32 -> bf16 conversion, 4 elements / thread (b128 in, b64 out)
// ---------------------------------------------------------------------------
__global__ void cvt_f32_bf16_kernel(const float* __restrict__ src,
                                    unsigned short* __restrict__ dst, int n4) {
    int i = blockIdx.x * blockDim.x + threadIdx.x;
    if (i < n4) {
        f32x4 v = *(const f32x4*)(src + 4 * (size_t)i);
        u16x4 o;
        o.x = f32_to_bf16(v.x);
        o.y = f32_to_bf16(v.y);
        o.z = f32_to_bf16(v.z);
        o.w = f32_to_bf16(v.w);
        *(u16x4*)(dst + 4 * (size_t)i) = o;
    }
}

// ---------------------------------------------------------------------------
// 2) GEMM: Y[M,N] = A[M,K] * W[N,K]^T   (y = x @ W.T, torch Linear layout)
//    Block = 256 thr (8 waves) -> 128 rows x 64 cols.
//    W tile (64x32 bf16 = 4KB) staged in double-buffered LDS via async copy;
//    each wave computes a 16x64 strip: 1 A-frag x 4 B-frags per K=32 step.
//    A-frag lane layout: chunks at k0+8*half and k0+16+8*half (16B each).
//    B-frag lane layout: chunks at 8*half and 16+8*half (16B each).
// ---------------------------------------------------------------------------
template <bool F32OUT>
__global__ __launch_bounds__(256)
void gemm_bf16_nt_kernel(const unsigned short* __restrict__ A,
                         const unsigned short* __restrict__ W,
                         float* __restrict__ outF,
                         unsigned short* __restrict__ outB,
                         const float* __restrict__ bias,
                         int M, int N, int K) {
    __shared__ unsigned short Wt[2][64][32];   // 8 KB double-buffered

    const int lane  = threadIdx.x & 31;
    const int wave  = threadIdx.x >> 5;
    const int row16 = lane & 15;
    const int half  = lane >> 4;

    const int nquad = N >> 6;                  // 64-wide col quads
    const int rb = blockIdx.x / nquad;         // 128-row block
    const int cq = blockIdx.x % nquad;

    // cooperative W-tile copy: thread -> one 16B chunk (64 rows x 4 chunks)
    const int ln = threadIdx.x >> 2;           // 0..63: W row within quad
    const int lk = (threadIdx.x & 3) * 8;      // 0,8,16,24: k chunk
    const unsigned short* wsrc = W + (size_t)(cq * 64 + ln) * K + lk;

    const unsigned short* arow =
        A + (size_t)(rb * 128 + wave * 16 + row16) * K;

    v8f acc[4] = {v8f{0}, v8f{0}, v8f{0}, v8f{0}};

    copy16_g2lds(wsrc, &Wt[0][ln][lk]);        // preload tile 0

    const int steps = K >> 5;
    for (int s = 0; s < steps; ++s) {
        const int buf = s & 1;
        wait_async_copies();
        __syncthreads();                       // tile s resident in Wt[buf]
        if (s + 1 < steps)
            copy16_g2lds(wsrc + (size_t)(s + 1) * 32, &Wt[buf ^ 1][ln][lk]);

        const int k0 = s * 32;
        Frag a;
        a.q[0] = *(const u32x4*)(arow + k0 + 8 * half);
        a.q[1] = *(const u32x4*)(arow + k0 + 16 + 8 * half);
        __builtin_prefetch(arow + k0 + 64, 0, 0);   // global_prefetch_b8

#pragma unroll
        for (int t = 0; t < 4; ++t) {
            const unsigned short* bp = &Wt[buf][t * 16 + row16][0];
            Frag b;
            b.q[0] = *(const u32x4*)(bp + 8 * half);
            b.q[1] = *(const u32x4*)(bp + 16 + 8 * half);
            acc[t] = wmma_bf16(a, b, acc[t]);
        }
    }

#pragma unroll
    for (int t = 0; t < 4; ++t) {
        const int col = cq * 64 + t * 16 + row16;
#pragma unroll
        for (int r = 0; r < 8; ++r) {
            const int row = rb * 128 + wave * 16 + r + 8 * half;
            float v = acc[t][r];
            if (F32OUT) {
                outF[(size_t)row * N + col] = v + bias[col];
            } else {
                outB[(size_t)row * N + col] = f32_to_bf16(v);
            }
        }
    }
}

// ---------------------------------------------------------------------------
// 3) V[b*N+m, h*64+d] (bf16) -> Vt[((b*H+h)*64+d), m] (bf16)
// ---------------------------------------------------------------------------
__global__ void transpose_v_kernel(const unsigned short* __restrict__ V,
                                   unsigned short* __restrict__ Vt) {
    int idx = blockIdx.x * blockDim.x + threadIdx.x;     // [b,h,d,m]
    int m   = idx & (NSEQ - 1);
    int r   = idx >> 10;
    int d   = r & (HDIM - 1);
    int r2  = r >> 6;
    int h   = r2 & (NHEADS - 1);
    int b   = r2 >> 4;
    Vt[idx] = V[(size_t)(b * NSEQ + m) * DIMC + h * HDIM + d];
}

// ---------------------------------------------------------------------------
// 4) Attention: per block = one (b, h, 16-row q-tile). 128 threads = 4 waves.
//    LDS: S[16][1024] f32 scores (64KB) + P[16][1024] bf16 probs (32KB).
// ---------------------------------------------------------------------------
__global__ __launch_bounds__(128)
void attention_kernel(const unsigned short* __restrict__ Q,
                      const unsigned short* __restrict__ Kmat,
                      const unsigned short* __restrict__ Vt,
                      const float* __restrict__ mask,
                      unsigned short* __restrict__ O) {
    extern __shared__ char smem[];
    float          (*S)[NSEQ] = (float(*)[NSEQ])smem;
    unsigned short (*P)[NSEQ] = (unsigned short(*)[NSEQ])(smem + 16 * NSEQ * sizeof(float));

    const int nt = blockIdx.x & 63;     // q-row tile (N/16 = 64)
    const int bh = blockIdx.x >> 6;
    const int h  = bh & (NHEADS - 1);
    const int b  = bh >> 4;
    const int n0 = nt * 16;

    const int lane  = threadIdx.x & 31;
    const int wave  = threadIdx.x >> 5;
    const int row16 = lane & 15;
    const int half  = lane >> 4;

    // Q tile 16x64 as two K=32 A-fragments
    const unsigned short* qrow = Q + (size_t)(b * NSEQ + n0 + row16) * DIMC + h * HDIM;
    Frag qa0, qa1;
    qa0.q[0] = *(const u32x4*)(qrow + 8 * half);
    qa0.q[1] = *(const u32x4*)(qrow + 16 + 8 * half);
    qa1.q[0] = *(const u32x4*)(qrow + 32 + 8 * half);
    qa1.q[1] = *(const u32x4*)(qrow + 48 + 8 * half);

    // --- S = Q K^T : each wave does 16 of the 64 m-tiles ---
    for (int mt = wave; mt < 64; mt += 4) {
        const unsigned short* krow =
            Kmat + (size_t)(b * NSEQ + mt * 16 + row16) * DIMC + h * HDIM;
        Frag kb0, kb1;
        kb0.q[0] = *(const u32x4*)(krow + 8 * half);
        kb0.q[1] = *(const u32x4*)(krow + 16 + 8 * half);
        kb1.q[0] = *(const u32x4*)(krow + 32 + 8 * half);
        kb1.q[1] = *(const u32x4*)(krow + 48 + 8 * half);
        v8f s = {0};
        s = wmma_bf16(qa0, kb0, s);
        s = wmma_bf16(qa1, kb1, s);
#pragma unroll
        for (int r = 0; r < 8; ++r)
            S[r + 8 * half][mt * 16 + row16] = s[r];
    }
    __syncthreads();

    // --- softmax over 1024 cols: 8 lanes per row, contiguous 128-col chunks,
    //     16B vector loads for both the mask stream and the LDS scores ---
    {
        const int srow = threadIdx.x >> 3;   // 0..15
        const int sub  = threadIdx.x & 7;    // 0..7 (contiguous, aligned in wave)
        const float* mrow =
            mask + ((size_t)b * NSEQ + (n0 + srow)) * NSEQ + sub * 128;
        float* sp = &S[srow][sub * 128];

        float mx = -3.4e38f;
        for (int c = 0; c < 128; c += 4) {
            f32x4 mv = *(const f32x4*)(mrow + c);
            f32x4 sv = *(f32x4*)(sp + c);
            sv = sv * ATTN_SCALE + mv;
            *(f32x4*)(sp + c) = sv;
            mx = fmaxf(mx, fmaxf(fmaxf(sv.x, sv.y), fmaxf(sv.z, sv.w)));
        }
        mx = fmaxf(mx, __shfl_xor(mx, 1, 32));
        mx = fmaxf(mx, __shfl_xor(mx, 2, 32));
        mx = fmaxf(mx, __shfl_xor(mx, 4, 32));

        float sum = 0.f;
        for (int c = 0; c < 128; ++c) {
            float e = __expf(sp[c] - mx);
            sp[c] = e;
            sum += e;
        }
        sum += __shfl_xor(sum, 1, 32);
        sum += __shfl_xor(sum, 2, 32);
        sum += __shfl_xor(sum, 4, 32);
        const float inv = 1.f / sum;

        unsigned short* pp = &P[srow][sub * 128];
        for (int c = 0; c < 128; c += 2) {
            unsigned int pk = (unsigned int)f32_to_bf16(sp[c] * inv) |
                              ((unsigned int)f32_to_bf16(sp[c + 1] * inv) << 16);
            *(unsigned int*)(pp + c) = pk;   // paired bf16 b32 store
        }
    }
    __syncthreads();

    // --- O = P * V : each wave owns one 16-col d-tile ---
    const unsigned short* vrow =
        Vt + (((size_t)b * NHEADS + h) * HDIM + wave * 16 + row16) * NSEQ;
    v8f acc = {0};
    for (int m0 = 0; m0 < NSEQ; m0 += 32) {
        Frag a, bfr;
        a.q[0]   = *(const u32x4*)&P[row16][m0 + 8 * half];
        a.q[1]   = *(const u32x4*)&P[row16][m0 + 16 + 8 * half];
        bfr.q[0] = *(const u32x4*)(vrow + m0 + 8 * half);
        bfr.q[1] = *(const u32x4*)(vrow + m0 + 16 + 8 * half);
        acc = wmma_bf16(a, bfr, acc);
    }
#pragma unroll
    for (int r = 0; r < 8; ++r)
        O[(size_t)(b * NSEQ + n0 + r + 8 * half) * DIMC + h * HDIM + wave * 16 + row16] =
            f32_to_bf16(acc[r]);
}

// ---------------------------------------------------------------------------
// Host launcher
// ---------------------------------------------------------------------------
extern "C" void kernel_launch(void* const* d_in, const int* in_sizes, int n_in,
                              void* d_out, int out_size, void* d_ws, size_t ws_size,
                              hipStream_t stream) {
    const float* x    = (const float*)d_in[0];
    const float* mask = (const float*)d_in[1];
    const float* Wq   = (const float*)d_in[2];
    const float* Wk   = (const float*)d_in[3];
    const float* Wv   = (const float*)d_in[4];
    const float* Wp   = (const float*)d_in[5];
    const float* bp   = (const float*)d_in[6];
    float* out        = (float*)d_out;

    const int M = BATCH * NSEQ;            // 4096 rows
    const size_t MB = 1024u * 1024u;
    char* ws = (char*)d_ws;
    unsigned short* xb  = (unsigned short*)(ws + 0 * MB);   // 8 MB
    unsigned short* Wqb = (unsigned short*)(ws + 8 * MB);   // 2 MB each
    unsigned short* Wkb = (unsigned short*)(ws + 10 * MB);
    unsigned short* Wvb = (unsigned short*)(ws + 12 * MB);
    unsigned short* Wpb = (unsigned short*)(ws + 14 * MB);
    unsigned short* Qb  = (unsigned short*)(ws + 16 * MB);  // 8 MB each
    unsigned short* Kb  = (unsigned short*)(ws + 24 * MB);
    unsigned short* Vb  = (unsigned short*)(ws + 32 * MB);
    unsigned short* Vt  = (unsigned short*)(ws + 40 * MB);
    unsigned short* Ob  = (unsigned short*)(ws + 48 * MB);

    // 1) converts (4 elem / thread)
    const int nx = M * DIMC, nw = DIMC * DIMC;
    cvt_f32_bf16_kernel<<<(nx / 4 + 255) / 256, 256, 0, stream>>>(x,  xb,  nx / 4);
    cvt_f32_bf16_kernel<<<(nw / 4 + 255) / 256, 256, 0, stream>>>(Wq, Wqb, nw / 4);
    cvt_f32_bf16_kernel<<<(nw / 4 + 255) / 256, 256, 0, stream>>>(Wk, Wkb, nw / 4);
    cvt_f32_bf16_kernel<<<(nw / 4 + 255) / 256, 256, 0, stream>>>(Wv, Wvb, nw / 4);
    cvt_f32_bf16_kernel<<<(nw / 4 + 255) / 256, 256, 0, stream>>>(Wp, Wpb, nw / 4);

    // 2) QKV projections: exact grid, 128 rows x 64 cols per block
    const int gblocks = (M / 128) * (DIMC / 64);   // 512
    gemm_bf16_nt_kernel<false><<<gblocks, 256, 0, stream>>>(xb, Wqb, nullptr, Qb, nullptr, M, DIMC, DIMC);
    gemm_bf16_nt_kernel<false><<<gblocks, 256, 0, stream>>>(xb, Wkb, nullptr, Kb, nullptr, M, DIMC, DIMC);
    gemm_bf16_nt_kernel<false><<<gblocks, 256, 0, stream>>>(xb, Wvb, nullptr, Vb, nullptr, M, DIMC, DIMC);

    // 3) V transpose to [b,h,d,m]
    transpose_v_kernel<<<(nx + 255) / 256, 256, 0, stream>>>(Vb, Vt);

    // 4) attention (96 KB dynamic LDS per workgroup)
    const int ablocks = BATCH * NHEADS * (NSEQ / 16);      // 4096
    const size_t lds  = 16 * NSEQ * sizeof(float) + 16 * NSEQ * sizeof(unsigned short);
    attention_kernel<<<ablocks, 128, lds, stream>>>(Qb, Kb, Vt, mask, Ob);

    // 5) output projection (+bias) -> fp32 d_out
    gemm_bf16_nt_kernel<true><<<gblocks, 256, 0, stream>>>(Ob, Wpb, out, nullptr, bp, M, DIMC, DIMC);
}
